// TransformerEncoder_25082609009270
// MI455X (gfx1250) — matrix operand order repaired and checked by reading, not compiled, not run on hardware
//
#include <hip/hip_runtime.h>
#include <math.h>

// ---------------------------------------------------------------------------
// MI455X (gfx1250) wave32 WMMA implementation, round 3.
// All matmuls run on v_wmma_f32_16x16x32_bf16 (bf16 operands, f32 accum).
// Operands pre-packed bf16 with K-pairs contiguous -> global_load_b128 pairs.
// k_conv3 register-tiled 2(cout-frag) x 4(pixel-frag): 1.5 loads/WMMA and
// 8 independent accumulators (no WMMA->WMMA hazard stalls).
// ---------------------------------------------------------------------------

typedef __attribute__((ext_vector_type(16))) __bf16 v16bf;
typedef __attribute__((ext_vector_type(8)))  float  v8f;

struct bfx2 { __bf16 x, y; };

static __device__ inline __bf16 f2bf(float f) { return (__bf16)f; }  // native cvt

static __device__ inline v8f wmma_bf16(v16bf a, v16bf b, v8f c) {
    return __builtin_amdgcn_wmma_f32_16x16x32_bf16(
        /*neg_a=*/false, a, /*neg_b=*/false, b,
        /*c_mod=*/(short)0, c, /*reuse_a=*/false, /*reuse_b=*/false);
}

#define ZERO8 {0.f,0.f,0.f,0.f,0.f,0.f,0.f,0.f}

// A-fragment k-index for slot i (contiguous k = 16*kh + s, matching frag_b)
static __device__ inline int a_k0(int i, int kh) {
    return kh * 16 + 2 * i;
}

// Build A fragment from row-major bf16 (element (m,k) = base[m*ld + k])
static __device__ inline v16bf frag_a(const __bf16* __restrict__ base, int ld, int m, int kh) {
    v16bf a;
    const __bf16* p = base + (size_t)m * ld;
#pragma unroll
    for (int i = 0; i < 8; ++i) {
        int k0 = a_k0(i, kh);
        bfx2 q = *(const bfx2*)(p + k0);
        a[2 * i] = q.x; a[2 * i + 1] = q.y;
    }
    return a;
}

// Build B fragment from n-major bf16 (element (k,n) = base[n*ld + k])
static __device__ inline v16bf frag_b(const __bf16* __restrict__ base, int ld, int n, int kh) {
    v16bf b;
    const __bf16* p = base + (size_t)n * ld + kh * 16;
#pragma unroll
    for (int i = 0; i < 8; ++i) {
        bfx2 q = *(const bfx2*)(p + 2 * i);
        b[2 * i] = q.x; b[2 * i + 1] = q.y;
    }
    return b;
}

// ---------------------------------------------------------------------------
// Pack kernels (run once per launch; trivially bandwidth-bound)
// ---------------------------------------------------------------------------

__global__ void k_cast(const float* __restrict__ src, __bf16* __restrict__ dst, int n) {
    int i = blockIdx.x * blockDim.x + threadIdx.x;
    if (i < n) dst[i] = f2bf(src[i]);
}

// (C,C,3,3) f32 -> [tap][cout][cin] bf16
__global__ void k_pack_w33(const float* __restrict__ src, __bf16* __restrict__ dst) {
    int i = blockIdx.x * blockDim.x + threadIdx.x;     // t*65536 + cout*256 + cin
    if (i >= 9 * 65536) return;
    int cin = i & 255, cout = (i >> 8) & 255, t = i >> 16;
    dst[i] = f2bf(src[((size_t)cout * 256 + cin) * 9 + t]);
}

// (B,C,H,W) f32 -> (B,H,W,C) bf16
__global__ void k_pack_nhwc(const float* __restrict__ src, __bf16* __restrict__ dst) {
    int i = blockIdx.x * blockDim.x + threadIdx.x;     // ((b*64+hh)*64+ww)*256 + c
    int c = i & 255;
    int ww = (i >> 8) & 63;
    int hh = (i >> 14) & 63;
    int b  = i >> 20;
    dst[i] = f2bf(src[(((size_t)b * 256 + c) * 64 + hh) * 64 + ww]);
}

// per-patch key mask bits: 1.0 if patch-mean(mask) < 0.5 else 0.0
__global__ void k_maskbits(const float* __restrict__ mask, float* __restrict__ bits) {
    int i = blockIdx.x * blockDim.x + threadIdx.x;     // 16*256
    if (i >= 16 * 256) return;
    int b = i >> 8, n = i & 255;
    int ph = n >> 4, pw = n & 15;
    float s = 0.f;
    for (int py = 0; py < 4; ++py)
        for (int px = 0; px < 4; ++px)
            s += mask[((size_t)b * 64 + (ph * 4 + py)) * 64 + (pw * 4 + px)];
    bits[i] = (s * (1.f / 16.f) < 0.5f) ? 1.f : 0.f;
}

// ---------------------------------------------------------------------------
// Kernel 1: q/k/v = 1x1 conv + bias + noise, patchified, stored bf16.
// One wave = one 16(cout) x 16(pixel) tile of one batch/one tensor.
// q,k: [b][head][n][d] ; v: [b][head][d][n] (transposed for PV B-operand)
// ---------------------------------------------------------------------------
__global__ void k_qkv(const __bf16* __restrict__ x16,   // NHWC bf16
                      const __bf16* __restrict__ wq16, const float* __restrict__ bq,
                      const __bf16* __restrict__ wk16, const float* __restrict__ bk,
                      const __bf16* __restrict__ wv16, const float* __restrict__ bv,
                      const float* __restrict__ nq, const float* __restrict__ nk,
                      const float* __restrict__ nv,
                      __bf16* __restrict__ qb, __bf16* __restrict__ kb,
                      __bf16* __restrict__ vb)
{
    int wid  = blockIdx.x * (blockDim.x >> 5) + (threadIdx.x >> 5);
    int lane = threadIdx.x & 31;
    int pt = wid & 255;  int t1 = wid >> 8;
    int ct = t1 & 15;    int t2 = t1 >> 4;
    int b  = t2 & 15;    int which = t2 >> 4;       // 0=q 1=k 2=v

    const __bf16* Wsel = which == 0 ? wq16 : (which == 1 ? wk16 : wv16);
    const float*  bsel = which == 0 ? bq   : (which == 1 ? bk   : bv);
    const float*  nsel = which == 0 ? nq   : (which == 1 ? nk   : nv);
    __bf16*       dsel = which == 0 ? qb   : (which == 1 ? kb   : vb);

    int hh = pt >> 2, ww0 = (pt & 3) << 4;
    int m = lane & 15, kh = lane >> 4, n = lane & 15;

    const __bf16* xb = x16 + ((size_t)(b * 64 + hh) * 64 + ww0) * 256;
    const __bf16* wa = Wsel + (size_t)ct * 16 * 256;
    v8f acc = ZERO8;

    for (int cs = 0; cs < 8; ++cs) {
        int cin0 = cs * 32;
        v16bf A  = frag_a(wa + cin0, 256, m, kh);
        v16bf Bf = frag_b(xb + cin0, 256, n, kh);
        acc = wmma_bf16(A, Bf, acc);
    }

#pragma unroll
    for (int r = 0; r < 8; ++r) {
        int cout = ct * 16 + r + 8 * kh;
        int ww = ww0 + n;
        int ph = hh >> 2, py = hh & 3, pw = ww >> 2, px = ww & 3;
        float val = acc[r] + bsel[cout]
                  + nsel[((((size_t)b * 256 + cout) * 16 + ph) * 4 + py) * 64 + pw * 4 + px];
        int h  = cout >> 4;
        int dd = (cout & 15) * 16 + py * 4 + px;
        int np = ph * 16 + pw;
        size_t bh = (size_t)(b * 16 + h);
        if (which == 2) dsel[(bh * 256 + dd) * 256 + np] = f2bf(val); // v transposed
        else            dsel[(bh * 256 + np) * 256 + dd] = f2bf(val);
    }
}

// ---------------------------------------------------------------------------
// Kernel 2: attention. One wave = 16 query rows of one (b,h).
// ---------------------------------------------------------------------------
#define AW 2   // waves per block
__global__ void k_attn(const __bf16* __restrict__ qb, const __bf16* __restrict__ kb,
                       const __bf16* __restrict__ vb, const float* __restrict__ dis,
                       const float* __restrict__ lap_a, const float* __restrict__ bits,
                       __bf16* __restrict__ attn16)
{
    __shared__ float  S[AW][16][256];
    __shared__ __align__(16) __bf16 P[AW][16][256];
    __shared__ float  redmax[AW][16][2];
    __shared__ float  redsum[AW][16][2];

    int w    = threadIdx.x >> 5;
    int lane = threadIdx.x & 31;
    int wid  = blockIdx.x * AW + w;
    int rt = wid & 15;  int h = (wid >> 4) & 15;  int b = wid >> 8;
    int n0 = rt * 16;
    size_t bh = (size_t)(b * 16 + h) * 65536;
    const __bf16* q = qb + bh;
    const __bf16* k = kb + bh;
    const __bf16* v = vb + bh;

    float la = lap_a[0];
    float sp = (la > 20.f) ? la : log1pf(__expf(la));

    int mrow = lane & 15, kh = lane >> 4, ncol = lane & 15;

    // hoist Q A-fragments (reused across all 16 key tiles)
    v16bf aq[8];
#pragma unroll
    for (int ks = 0; ks < 8; ++ks)
        aq[ks] = frag_a(q + (size_t)n0 * 256 + ks * 32, 256, mrow, kh);

    // ---- scores ----
    for (int mt = 0; mt < 16; ++mt) {
        v8f acc = ZERO8;
#pragma unroll
        for (int ks = 0; ks < 8; ++ks)
            acc = wmma_bf16(aq[ks], frag_b(k + (size_t)mt * 16 * 256 + ks * 32, 256, ncol, kh), acc);
#pragma unroll
        for (int r = 0; r < 8; ++r) {
            int srow = r + 8 * kh;
            int gm = mt * 16 + ncol, gn = n0 + srow;
            float s = (acc[r] + sp * dis[(size_t)gn * 256 + gm]) * 0.0625f;
            if (bits[b * 256 + gm] != 0.f) s = -__builtin_inff();
            S[w][srow][gm] = s;
        }
    }
    __syncthreads();

    // ---- softmax: lane -> (row = lane>>1, half-row of 128) ----
    {
        int row = lane >> 1, half = lane & 1, c0 = half * 128;
        float mx = -__builtin_inff();
        for (int j = 0; j < 128; ++j) mx = fmaxf(mx, S[w][row][c0 + j]);
        redmax[w][row][half] = mx;
        __syncthreads();
        float rmx = fmaxf(redmax[w][row][0], redmax[w][row][1]);
        float sum = 0.f;
        for (int j = 0; j < 128; ++j) {
            float e = __expf(S[w][row][c0 + j] - rmx);
            S[w][row][c0 + j] = e;
            sum += e;
        }
        redsum[w][row][half] = sum;
        __syncthreads();
        float inv = 1.f / (redsum[w][row][0] + redsum[w][row][1]);
        for (int j = 0; j < 128; ++j)
            P[w][row][c0 + j] = f2bf(S[w][row][c0 + j] * inv);
    }
    __syncthreads();

    // ---- PV ----
    v16bf ap[8];
#pragma unroll
    for (int ks = 0; ks < 8; ++ks)
        ap[ks] = frag_a(&P[w][0][ks * 32], 256, mrow, kh);

    for (int dt = 0; dt < 16; ++dt) {
        v8f acc = ZERO8;
#pragma unroll
        for (int ks = 0; ks < 8; ++ks)
            acc = wmma_bf16(ap[ks], frag_b(v + (size_t)dt * 16 * 256 + ks * 32, 256, ncol, kh), acc);
#pragma unroll
        for (int r = 0; r < 8; ++r) {
            int srow = r + 8 * kh;
            int gn = n0 + srow, gd = dt * 16 + ncol;
            int c = h * 16 + (gd >> 4);
            int pix = gd & 15;
            int hh = (gn >> 4) * 4 + (pix >> 2);
            int ww = (gn & 15) * 4 + (pix & 3);
            attn16[((size_t)(b * 64 + hh) * 64 + ww) * 256 + c] = f2bf(acc[r]);
        }
    }
}

// ---------------------------------------------------------------------------
// Kernel 3: 3x3 conv (dilation param) as implicit GEMM, register-tiled.
// One wave = 32(cout) x 64(pixel = one image row).  Per (tap, K-step):
// 2 A-fragments x 4 B-fragments -> 8 independent WMMAs (1.5 b128 loads/WMMA).
// resid != nullptr : val = lrelu(acc+bias) + resid   (W_out path)
// dst32 (f32 NCHW) and dst16 (bf16 NHWC) each optional.
// ---------------------------------------------------------------------------
__global__ void k_conv3(const __bf16* __restrict__ src16, const __bf16* __restrict__ wt16,
                        const float* __restrict__ bias, const float* __restrict__ resid,
                        float* __restrict__ dst32, __bf16* __restrict__ dst16, int dil)
{
    int wid  = blockIdx.x * (blockDim.x >> 5) + (threadIdx.x >> 5);
    int lane = threadIdx.x & 31;
    int hh  = wid & 63;                 // image row
    int ct2 = (wid >> 6) & 7;           // cout group of 32
    int b   = wid >> 9;
    int m = lane & 15, kh = lane >> 4, n = lane & 15;

    v8f acc[2][4] = { {ZERO8, ZERO8, ZERO8, ZERO8}, {ZERO8, ZERO8, ZERO8, ZERO8} };

    for (int t = 0; t < 9; ++t) {
        int ky = t / 3, kx = t % 3;
        int yy = hh + (ky - 1) * dil;
        int dx = (kx - 1) * dil;
        bool yok = ((unsigned)yy < 64u);
        const __bf16* srow = src16 + (size_t)(b * 64 + yy) * 64 * 256;
        const __bf16* wb   = wt16 + ((size_t)t * 256 + ct2 * 32) * 256;
        for (int cs = 0; cs < 8; ++cs) {
            int cin0 = cs * 32;
            v16bf A0 = frag_a(wb + cin0,            256, m, kh);
            v16bf A1 = frag_a(wb + 16 * 256 + cin0, 256, m, kh);
            v16bf Bg[4];
#pragma unroll
            for (int g = 0; g < 4; ++g) {
                int xx = g * 16 + n + dx;
                bool ok = yok && ((unsigned)xx < 64u);
                if (ok) {
                    const __bf16* p = srow + (size_t)xx * 256 + cin0 + kh * 16;
#pragma unroll
                    for (int i = 0; i < 8; ++i) {
                        bfx2 qq = *(const bfx2*)(p + 2 * i);
                        Bg[g][2 * i] = qq.x; Bg[g][2 * i + 1] = qq.y;
                    }
                } else {
#pragma unroll
                    for (int i = 0; i < 16; ++i) Bg[g][i] = (__bf16)0.f;
                }
            }
#pragma unroll
            for (int g = 0; g < 4; ++g) {
                acc[0][g] = wmma_bf16(A0, Bg[g], acc[0][g]);
                acc[1][g] = wmma_bf16(A1, Bg[g], acc[1][g]);
            }
        }
    }

#pragma unroll
    for (int c = 0; c < 2; ++c)
#pragma unroll
    for (int g = 0; g < 4; ++g)
#pragma unroll
    for (int r = 0; r < 8; ++r) {
        int cout = ct2 * 32 + c * 16 + r + 8 * kh;
        int ww = g * 16 + n;
        size_t idx = (((size_t)b * 256 + cout) * 64 + hh) * 64 + ww;   // NCHW
        float val = acc[c][g][r] + bias[cout];
        if (resid) val = (val >= 0.f ? val : 0.2f * val) + resid[idx];
        if (dst32) dst32[idx] = val;
        if (dst16) dst16[((size_t)(b * 64 + hh) * 64 + ww) * 256 + cout] = f2bf(val);
    }
}

// ---------------------------------------------------------------------------
// Kernel 4: instance-norm + leaky-relu.  One block per (b,c) plane (NCHW src).
// ---------------------------------------------------------------------------
__global__ void k_innorm(const float* __restrict__ src, const float* __restrict__ addsrc,
                         float* __restrict__ dst32, __bf16* __restrict__ dst16)
{
    __shared__ float s1[256], s2[256];
    int bc = blockIdx.x, tid = threadIdx.x;
    int b = bc >> 8, c = bc & 255;
    const float* p = src + (size_t)bc * 4096;
    float sum = 0.f, sq = 0.f;
    for (int i = tid; i < 4096; i += 256) { float x = p[i]; sum += x; sq += x * x; }
    s1[tid] = sum; s2[tid] = sq; __syncthreads();
    for (int off = 128; off > 0; off >>= 1) {
        if (tid < off) { s1[tid] += s1[tid + off]; s2[tid] += s2[tid + off]; }
        __syncthreads();
    }
    float mu  = s1[0] * (1.f / 4096.f);
    float var = s2[0] * (1.f / 4096.f) - mu * mu;
    float rs  = rsqrtf(var + 1e-5f);
    for (int i = tid; i < 4096; i += 256) {
        float x = (p[i] - mu) * rs;
        x = x >= 0.f ? x : 0.2f * x;
        size_t idx = (size_t)bc * 4096 + i;
        if (dst32) dst32[idx] = addsrc ? (addsrc[idx] + x) : x;
        if (dst16) dst16[((size_t)b * 4096 + i) * 256 + c] = f2bf(x);
    }
}

// ---------------------------------------------------------------------------
// Host launcher
// ---------------------------------------------------------------------------
extern "C" void kernel_launch(void* const* d_in, const int* in_sizes, int n_in,
                              void* d_out, int out_size, void* d_ws, size_t ws_size,
                              hipStream_t stream)
{
    const float* x    = (const float*)d_in[0];
    const float* mask = (const float*)d_in[1];
    const float* dis  = (const float*)d_in[2];
    const float* lap  = (const float*)d_in[3];
    const float* Wq = (const float*)d_in[4];  const float* bq = (const float*)d_in[5];
    const float* Wk = (const float*)d_in[6];  const float* bk = (const float*)d_in[7];
    const float* Wv = (const float*)d_in[8];  const float* bv = (const float*)d_in[9];
    const float* nq = (const float*)d_in[10];
    const float* nk = (const float*)d_in[11];
    const float* nv = (const float*)d_in[12];
    const float* Wo = (const float*)d_in[13]; const float* bo = (const float*)d_in[14];
    const float* W1 = (const float*)d_in[15]; const float* b1 = (const float*)d_in[16];
    const float* W2 = (const float*)d_in[17]; const float* b2 = (const float*)d_in[18];
    float* out = (float*)d_out;

    // ---- workspace layout (~239 MB), with live-range reuse ----
    // A [0,100.7MB):    q/k/v bf16 -> later t1 f32 (64MB) + t1n16 bf16 (33.5MB)
    // B [100.7,134.2):  xb16 NHWC bf16 -> attn16 bf16 -> (with D) t2 f32 (64MB)
    // D [134.2,167.8):  outb16 NHWC bf16 (t2 spans B+D after it is dead)
    // C [167.8,234.9):  outb f32 NCHW (live to the end)
    // W [234.9,...):    bf16 weights + mask bits
    char* ws = (char*)d_ws;
    __bf16* qb     = (__bf16*)(ws);
    __bf16* kb     = (__bf16*)(ws + 33554432u);
    __bf16* vb     = (__bf16*)(ws + 67108864u);
    float*  t1     = (float*)(ws);                       // after qkv dead
    __bf16* t1n16  = (__bf16*)(ws + 67108864u);
    __bf16* xb16   = (__bf16*)(ws + 100663296u);
    __bf16* attn16 = (__bf16*)(ws + 100663296u);         // after xb16 dead
    float*  t2     = (float*)(ws + 100663296u);          // spans B+D after both dead
    __bf16* outb16 = (__bf16*)(ws + 134217728u);
    float*  outb   = (float*)(ws + 167772160u);
    __bf16* wq16   = (__bf16*)(ws + 234881024u);
    __bf16* wk16   = (__bf16*)(ws + 235012096u);
    __bf16* wv16   = (__bf16*)(ws + 235143168u);
    __bf16* wo16   = (__bf16*)(ws + 235274240u);
    __bf16* w116   = (__bf16*)(ws + 236453888u);
    __bf16* w216   = (__bf16*)(ws + 237633536u);
    float*  bits   = (float*)(ws + 238813184u);

    // ---- pack passes ----
    k_maskbits <<<16, 256, 0, stream>>>(mask, bits);
    k_cast     <<<256, 256, 0, stream>>>(Wq, wq16, 65536);
    k_cast     <<<256, 256, 0, stream>>>(Wk, wk16, 65536);
    k_cast     <<<256, 256, 0, stream>>>(Wv, wv16, 65536);
    k_pack_w33 <<<2304, 256, 0, stream>>>(Wo, wo16);
    k_pack_w33 <<<2304, 256, 0, stream>>>(W1, w116);
    k_pack_w33 <<<2304, 256, 0, stream>>>(W2, w216);
    k_pack_nhwc<<<65536, 256, 0, stream>>>(x, xb16);

    // ---- main pipeline ----
    k_qkv    <<<24576, 256, 0, stream>>>(xb16, wq16, bq, wk16, bk, wv16, bv,
                                         nq, nk, nv, qb, kb, vb);
    k_attn   <<<2048, 32 * AW, 0, stream>>>(qb, kb, vb, dis, lap, bits, attn16);
    k_conv3  <<<1024, 256, 0, stream>>>(attn16, wo16, bo, x,  outb, outb16, 1); // lrelu+resid
    k_conv3  <<<1024, 256, 0, stream>>>(outb16, w116, b1, nullptr, t1, nullptr, 2);
    k_innorm <<<4096, 256, 0, stream>>>(t1, nullptr, nullptr, t1n16);           // IN+lrelu -> bf16 NHWC
    k_conv3  <<<1024, 256, 0, stream>>>(t1n16, w216, b2, nullptr, t2, nullptr, 1);
    k_innorm <<<4096, 256, 0, stream>>>(t2, outb, out, nullptr);                // out + lrelu(IN(t2))
}
